// StaticGraphModule_53790170415315
// MI455X (gfx1250) — matrix operand order repaired and checked by reading, not compiled, not run on hardware
//
#include <hip/hip_runtime.h>

// CDNA5 / gfx1250: wave32, WMMA bf16 -> f32 accumulate, async global->LDS staging
typedef __attribute__((ext_vector_type(16))) __bf16 v16bf;
typedef __attribute__((ext_vector_type(8)))  __bf16 v8bf;
typedef __attribute__((ext_vector_type(4)))  __bf16 v4bf;
typedef __attribute__((ext_vector_type(2)))  __bf16 v2bf;
typedef __attribute__((ext_vector_type(8)))  float  v8f;

#define Bn 4
#define Cn 64
#define Hn 512
#define Wn 512
#define TILE_W 64
#define XS_S 72     // f32 stride per (row,channel) line: [3 halo-L][64 interior][halo-R][pad]
#define WL_S 136    // bf16 stride per Wp row (128 + 8 pad -> lane banks n*4 mod 64, distinct)
#define TB_S 72     // bf16 stride per transposed pixel column (lane banks n*36 mod 64, distinct)
#define NTHREADS 128

static __device__ __forceinline__ v16bf cat8(v8bf lo, v8bf hi) {
    return __builtin_shufflevector(lo, hi, 0,1,2,3,4,5,6,7,8,9,10,11,12,13,14,15);
}

// Async 16B global->LDS copy (GLOBAL_LOAD_ASYNC_TO_LDS_B128, tracked by ASYNCcnt).
// lds_off: LDS byte address (low 32 bits of generic pointer, per LDS aperture rule).
// base+goff: GVS addressing (SGPR 64-bit base + per-lane 32-bit byte offset).
static __device__ __forceinline__ void async_g2l_b128(unsigned lds_off,
                                                      unsigned goff,
                                                      const float* base) {
    asm volatile("global_load_async_to_lds_b128 %0, %1, %2"
                 :: "v"(lds_off), "v"(goff), "s"(base)
                 : "memory");
}
static __device__ __forceinline__ void wait_async0() {
    asm volatile("s_wait_asynccnt 0x0" ::: "memory");
}

__global__ __launch_bounds__(NTHREADS)
void neighbor_mix_wmma_kernel(const float* __restrict__ x,
                              const float* __restrict__ Wp,   // (64, 128) row-major
                              const float* __restrict__ bias, // (64,)
                              float* __restrict__ out)
{
    __shared__ float  xs  [3 * Cn * XS_S];   // 55296 B : 3 rows x 64 ch x (halo+64+halo) f32
    __shared__ __bf16 wlds[Cn * WL_S];       // 17408 B : Wp in bf16, row-major padded
    __shared__ __bf16 xmb [TILE_W * TB_S];   //  9216 B : center x, transposed [pixel][ch] bf16
    __shared__ __bf16 mnb [TILE_W * TB_S];   //  9216 B : 8-neighbor mean, transposed bf16

    const int tid = threadIdx.x;
    const int w0  = blockIdx.x * TILE_W;
    const int h   = blockIdx.y;
    const int bb  = blockIdx.z;

    // ---- Phase 1a: interior columns of 3 rows (clamped), async 16B global->LDS ----
    for (int i = tid; i < 3 * Cn * 16; i += NTHREADS) {
        int r3 = i >> 10, rem = i & 1023;
        int c = rem >> 4, v = rem & 15;
        int hs = h + r3 - 1; hs = hs < 0 ? 0 : (hs > Hn - 1 ? Hn - 1 : hs);
        unsigned goff = (unsigned)((((unsigned)bb * Cn + c) * Hn + hs) * Wn + w0 + 4u * v) * 4u;
        unsigned loff = (unsigned)(uintptr_t)&xs[(r3 * Cn + c) * XS_S + 4 + 4 * v];
        async_g2l_b128(loff, goff, x);       // 16B, both sides 16B-aligned
    }
    // ---- Phase 1b: halo columns (clamp-to-edge), regular loads ----
    for (int j = tid; j < 3 * Cn * 2; j += NTHREADS) {
        int r3 = j / (Cn * 2), rem = j % (Cn * 2);
        int c = rem >> 1, side = rem & 1;
        int hs = h + r3 - 1; hs = hs < 0 ? 0 : (hs > Hn - 1 ? Hn - 1 : hs);
        int ws = side ? (w0 + TILE_W < Wn ? w0 + TILE_W : Wn - 1)
                      : (w0 > 0 ? w0 - 1 : 0);
        xs[(r3 * Cn + c) * XS_S + (side ? 4 + TILE_W : 3)] =
            x[(((size_t)bb * Cn + c) * Hn + hs) * Wn + ws];
    }
    // ---- Phase 1c: Wp f32 -> bf16 into LDS, once per workgroup ----
    const float4* wv4 = (const float4*)Wp;
    for (int i = tid; i < Cn * 32; i += NTHREADS) {
        int o = i >> 5, v = i & 31;
        float4 wq = wv4[o * 32 + v];
        v4bf t; t[0] = (__bf16)wq.x; t[1] = (__bf16)wq.y;
                t[2] = (__bf16)wq.z; t[3] = (__bf16)wq.w;
        *(v4bf*)&wlds[o * WL_S + 4 * v] = t;                   // 8B-aligned
    }
    wait_async0();        // ASYNCcnt == 0: all async LDS writes landed
    __syncthreads();

    // ---- Phase 2: cooperative mean + transpose to [pixel][channel] bf16 panels ----
    for (int i = tid; i < TILE_W * 32; i += NTHREADS) {
        int col = i >> 5, c0 = (i & 31) * 2;
        float m[2], xc[2];
        #pragma unroll
        for (int u = 0; u < 2; ++u) {
            const float* r0 = &xs[(0 * Cn + c0 + u) * XS_S + 4 + col];
            const float* r1 = &xs[(1 * Cn + c0 + u) * XS_S + 4 + col];
            const float* r2 = &xs[(2 * Cn + c0 + u) * XS_S + 4 + col];
            m[u]  = (r0[-1] + r0[0] + r0[1]
                   + r1[-1]         + r1[1]
                   + r2[-1] + r2[0] + r2[1]) * 0.125f;
            xc[u] = r1[0];
        }
        v2bf tm; tm[0] = (__bf16)m[0];  tm[1] = (__bf16)m[1];
        v2bf tx; tx[0] = (__bf16)xc[0]; tx[1] = (__bf16)xc[1];
        *(v2bf*)&mnb[col * TB_S + c0] = tm;                    // 4B-aligned (c0 even)
        *(v2bf*)&xmb[col * TB_S + c0] = tx;
    }
    __syncthreads();

    // ---- Phase 3: per-wave WMMA. 16 pixels/wave, M=64 (4 tiles), K=128 (4 tiles) ----
    const int wv_  = tid >> 5;     // wave id 0..3
    const int lane = tid & 31;
    const int n    = lane & 15;    // N (pixel) for B/D, M row for A
    const int g    = lane >> 4;    // half-select
    const int px   = wv_ * 16;
    const int col  = px + n;       // this lane's pixel column for B/D

    // B operands: K-contiguous bf16 -> two aligned 16B LDS loads per tile.
    // Lane layout: element e -> K = 16*g + e  (K-tile kt covers channels kt*32..kt*32+31)
    v16bf Bt[4];
    #pragma unroll
    for (int kt = 0; kt < 2; ++kt) {          // K = 0..63  : x channels
        const __bf16* p = &xmb[col * TB_S + kt * 32 + g * 16];
        Bt[kt] = cat8(*(const v8bf*)p, *(const v8bf*)(p + 8));
    }
    #pragma unroll
    for (int kt = 0; kt < 2; ++kt) {          // K = 64..127 : mean channels
        const __bf16* p = &mnb[col * TB_S + kt * 32 + g * 16];
        Bt[2 + kt] = cat8(*(const v8bf*)p, *(const v8bf*)(p + 8));
    }

    #pragma unroll
    for (int mt = 0; mt < 4; ++mt) {
        v8f acc = {};
        #pragma unroll
        for (int kt = 0; kt < 4; ++kt) {
            // A layout: lane m = lane&15; element e -> K = kt*32 + 16g + e
            const __bf16* wr = &wlds[(mt * 16 + n) * WL_S + kt * 32 + g * 16];
            v16bf At = cat8(*(const v8bf*)wr, *(const v8bf*)(wr + 8));
            acc = __builtin_amdgcn_wmma_f32_16x16x32_bf16(
                      /*neg_a=*/false, At, /*neg_b=*/false, Bt[kt],
                      /*c_mod=*/(short)0, acc, /*reuse_a=*/false, /*reuse_b=*/false);
        }
        // Epilogue: bias + ReLU + f32 residual, coalesced stores
        #pragma unroll
        for (int r = 0; r < 8; ++r) {
            int   o = mt * 16 + g * 8 + r;                   // D layout: M = r + 8*g
            float v = acc[r] + bias[o];
            v = v > 0.0f ? v : 0.0f;
            v += xs[(Cn + o) * XS_S + 4 + col];              // residual x[o, pixel] (f32)
            out[(((size_t)bb * Cn + o) * Hn + h) * Wn + (w0 + col)] = v;
        }
    }
}

extern "C" void kernel_launch(void* const* d_in, const int* in_sizes, int n_in,
                              void* d_out, int out_size, void* d_ws, size_t ws_size,
                              hipStream_t stream) {
    const float* x    = (const float*)d_in[0];  // (4, 64, 512, 512)
    const float* Wp   = (const float*)d_in[1];  // (64, 128)
    const float* bias = (const float*)d_in[2];  // (64,)
    float*       out  = (float*)d_out;          // (4, 64, 512, 512)

    dim3 grid(Wn / TILE_W, Hn, Bn);             // (8, 512, 4)
    dim3 block(NTHREADS);
    neighbor_mix_wmma_kernel<<<grid, block, 0, stream>>>(x, Wp, bias, out);
}
